// Model_26130581029007
// MI455X (gfx1250) — matrix operand order repaired and checked by reading, not compiled, not run on hardware
//
#include <hip/hip_runtime.h>
#include <math.h>

// ---------------- dims ----------------
#define B_    32
#define S_    512
#define ENC_  862
#define ENCP_ 864            // ENC padded to multiple of 32 (K-dim of GEMMs)
#define DM_   512
#define H_    8
#define HD_   64
#define DFF_  2048
#define P_    96
#define L_    2
#define BSROWS (B_*S_)       // 16384

// ---------------- WMMA types ----------------
typedef __attribute__((ext_vector_type(16))) __bf16 bf16x16;
typedef __attribute__((ext_vector_type(8)))  float  v8f;
typedef __attribute__((ext_vector_type(4)))  unsigned int u32x4;
typedef __attribute__((ext_vector_type(8)))  int i32x8;
typedef __attribute__((ext_vector_type(4)))  int i32x4;

union FragBF { bf16x16 v; unsigned int w[8]; unsigned short u[16]; };

__device__ __forceinline__ unsigned short f2bf(float f) {
  unsigned int x = __builtin_bit_cast(unsigned int, f);
  x += 0x7FFFu + ((x >> 16) & 1u);            // round-to-nearest-even
  return (unsigned short)(x >> 16);
}
// 16-bit A-matrix 16x32 K index for VGPR j, lane-half h (contiguous: k = 16h + 2j)
__device__ __forceinline__ int aK(int j, int half) {
  return 16*half + 2*j;
}
__device__ __forceinline__ v8f v8zero() {
  v8f z;
#pragma unroll
  for (int i = 0; i < 8; ++i) z[i] = 0.f;
  return z;
}

// ---------------- Tensor Data Mover: 2-D tile global->LDS (ISA ch. 8) ----------------
// D# group0: count=1 | lds_addr | global_addr(57b) | type=2
// D# group1: data_size=2B, pad_enable, pad_interval/amount, dims/strides (2-D)
__device__ __forceinline__ void tdm_load_2d(unsigned lds_addr, const void* gptr,
    unsigned tile_d0, unsigned tile_d1, unsigned tensor_d0, unsigned tensor_d1,
    unsigned stride0_elems, unsigned pad_interval_code, unsigned pad_amount_code) {
  unsigned long long ga = (unsigned long long)(size_t)gptr;
  u32x4 g0;
  g0[0] = 1u;                                              // count=1 (valid user D#)
  g0[1] = lds_addr;                                        // LDS byte address
  g0[2] = (unsigned)(ga & 0xFFFFFFFFu);                    // global_addr[31:0]
  g0[3] = (unsigned)((ga >> 32) & 0x1FFFFFFu) | (2u << 30);// global_addr[56:32] | type=2
  i32x8 g1;
  g1[0] = (int)((1u << 16)                                  // data_size = 2 bytes
              | (1u << 20)                                  // pad_enable
              | (pad_interval_code << 22)
              | (pad_amount_code << 25));
  g1[1] = (int)((tensor_d0 & 0xFFFFu) << 16);               // tensor_dim0[15:0]
  g1[2] = (int)(((tensor_d0 >> 16) & 0xFFFFu) | ((tensor_d1 & 0xFFFFu) << 16));
  g1[3] = (int)(((tensor_d1 >> 16) & 0xFFFFu) | (tile_d0 << 16));   // tile_dim0
  g1[4] = (int)(tile_d1 & 0xFFFFu);                         // tile_dim1 (tile_dim2=0)
  g1[5] = (int)stride0_elems;                               // tensor_dim0_stride[31:0]
  g1[6] = 0;
  g1[7] = 0;
  i32x4 z4; z4[0] = 0; z4[1] = 0; z4[2] = 0; z4[3] = 0;
  i32x8 z8;
#pragma unroll
  for (int i = 0; i < 8; ++i) z8[i] = 0;
#if __has_builtin(__builtin_amdgcn_tensor_load_to_lds)
  __builtin_amdgcn_tensor_load_to_lds(g0, g1, z4, z4, z8, 0);
#endif
}
__device__ __forceinline__ void tdm_wait() {
#if __has_builtin(__builtin_amdgcn_s_wait_tensorcnt)
  __builtin_amdgcn_s_wait_tensorcnt(0);
#endif
}

// ---------------- elementwise / conversion kernels ----------------
__global__ void k_cvt_bf16(const float* __restrict__ s, unsigned short* __restrict__ d, int n) {
  int i = blockIdx.x * 256 + threadIdx.x;
  if (i < n) d[i] = f2bf(s[i]);
}

// [K,N] f32 -> [KP,N] bf16 with zero rows K..KP
__global__ void k_cvt_pad(const float* __restrict__ s, unsigned short* __restrict__ d,
                          int K, int N, int KP) {
  int i = blockIdx.x * 256 + threadIdx.x;
  if (i >= KP * N) return;
  int r = i / N, c = i % N;
  d[i] = (r < K) ? f2bf(s[(size_t)r * N + c]) : (unsigned short)0;
}

// wq/wk/wv [H,ENC,HD] -> row-major [ENCP, H*HD] (zero pad rows)
__global__ void k_cvt_qkvw(const float* __restrict__ w, unsigned short* __restrict__ d, int n) {
  int i = blockIdx.x * 256 + threadIdx.x;
  if (i >= n) return;
  int c = i % DM_, e = i / DM_;
  int h = c / HD_, dd = c % HD_;
  d[i] = (e < ENC_) ? f2bf(w[((size_t)h * ENC_ + e) * HD_ + dd]) : (unsigned short)0;
}

__global__ __launch_bounds__(256) void k_layernorm(const float* __restrict__ x,
    const float* __restrict__ w, const float* __restrict__ b,
    unsigned short* __restrict__ y, int C, int CP) {
  __shared__ float r1[256], r2[256];
  const int row = blockIdx.x;
  const float* xr = x + (size_t)row * C;
  float s1 = 0.f, s2 = 0.f;
  for (int c = threadIdx.x; c < C; c += 256) { float v = xr[c]; s1 += v; s2 += v * v; }
  r1[threadIdx.x] = s1; r2[threadIdx.x] = s2;
  __syncthreads();
  for (int off = 128; off > 0; off >>= 1) {
    if (threadIdx.x < off) { r1[threadIdx.x] += r1[threadIdx.x + off]; r2[threadIdx.x] += r2[threadIdx.x + off]; }
    __syncthreads();
  }
  float m  = r1[0] / (float)C;
  float var = r2[0] / (float)C - m * m;
  float rs = rsqrtf(var + 1e-5f);
  for (int c = threadIdx.x; c < CP; c += 256) {
    float v = (c < C) ? (xr[c] - m) * rs * w[c] + b[c] : 0.f;
    y[(size_t)row * CP + c] = f2bf(v);
  }
}

__global__ void k_gate(const float* __restrict__ G, const float* __restrict__ Yn,
                       unsigned short* __restrict__ o, int n) {
  int i = blockIdx.x * 256 + threadIdx.x;
  if (i >= n) return;
  float g = G[i];
  float sw = g / (1.f + expf(-g));           // swish
  o[i] = f2bf(sw * Yn[i]);
}

__global__ void k_mul(const float* __restrict__ a, const float* __restrict__ b,
                      unsigned short* __restrict__ o, int n) {
  int i = blockIdx.x * 256 + threadIdx.x;
  if (i < n) o[i] = f2bf(a[i] * b[i]);
}

// X [B,S,ENC] f32 -> Xt [B,ENC,S] bf16
__global__ void k_transpose_cvt(const float* __restrict__ X, unsigned short* __restrict__ Xt, int n) {
  int i = blockIdx.x * 256 + threadIdx.x;
  if (i >= n) return;
  int s = i % S_;
  int e = (i / S_) % ENC_;
  int b = i / (S_ * ENC_);
  Xt[i] = f2bf(X[((size_t)b * S_ + s) * ENC_ + e]);
}

__global__ void k_gelu_bias(const float* __restrict__ h, const float* __restrict__ bias,
                            unsigned short* __restrict__ o, int n, int C) {
  int i = blockIdx.x * 256 + threadIdx.x;
  if (i >= n) return;
  float x = h[i] + bias[i % C];
  float g = 0.5f * x * (1.f + erff(x * 0.70710678118654752f));  // exact gelu
  o[i] = f2bf(g);
}

// H2 [bc,ENC,P] f32 -> out [bc,P,ENC] (+bias over P)
__global__ void k_out_bias_tr(const float* __restrict__ H2, const float* __restrict__ b2,
                              float* __restrict__ out, int n) {
  int i = blockIdx.x * 256 + threadIdx.x;
  if (i >= n) return;
  int p = i % P_;
  int e = (i / P_) % ENC_;
  int bb = i / (P_ * ENC_);
  out[((size_t)bb * P_ + p) * ENC_ + e] = H2[i] + b2[p];
}

// ---------------- tiled bf16 WMMA GEMM ----------------
// C[M,N] = A[M,K] (bf16 row-major, K % 32 == 0) x W[K,N] (bf16 row-major) [+ addsrc]
// block 256 = 8 waves; tile 128x64, BK=32; wave w -> rows [16w,16w+16), 4 col tiles.
// Fast path (full M/N tile): TDM loads the A tile (LDS pad -> stride 40), B unguarded.
#define ASTR 40
__global__ __launch_bounds__(256) void k_gemm(const unsigned short* __restrict__ A,
    const unsigned short* __restrict__ W, float* __restrict__ Cf,
    unsigned short* __restrict__ Cb, const float* __restrict__ addsrc,
    int M, int N, int K) {
  __shared__ unsigned short As[128 * ASTR];   // [m][k], stride 40 (TDM pad: 16dw + 4dw)
  __shared__ unsigned short Bs[64 * 34];      // [n][k], stride 34
  const int tid = threadIdx.x, lane = tid & 31, wave = tid >> 5;
  const int l15 = lane & 15, half = lane >> 4;
  const int m0 = blockIdx.y * 128, n0 = blockIdx.x * 64;
  const bool fast = (m0 + 128 <= M) && (n0 + 64 <= N);
  const unsigned asBase = (unsigned)(size_t)As;

  v8f acc[4];
#pragma unroll
  for (int t = 0; t < 4; ++t) acc[t] = v8zero();

  for (int k0 = 0; k0 < K; k0 += 32) {
    if (fast) {
      if (wave == 0)   // one TDM op moves the whole 128x32 tile; pad -> stride 40
        tdm_load_2d(asBase, A + (size_t)m0 * K + k0, 32u, 128u,
                    (unsigned)(K - k0), (unsigned)(M - m0), (unsigned)K, 3u, 3u);
      // B tile transpose -> [n][k], unguarded
#pragma unroll
      for (int it = 0; it < 4; ++it) {
        int e = tid + it * 256;
        int kk = e >> 5, n2 = (e & 31) * 2;
        unsigned int v = *(const unsigned int*)&W[(size_t)(k0 + kk) * N + (n0 + n2)];
        Bs[n2 * 34 + kk]       = (unsigned short)(v & 0xFFFFu);
        Bs[(n2 + 1) * 34 + kk] = (unsigned short)(v >> 16);
      }
      if (wave == 0) tdm_wait();
    } else {
#pragma unroll
      for (int it = 0; it < 8; ++it) {
        int e = tid + it * 256;
        int r = e >> 4, c2 = (e & 15) * 2;
        int gr = m0 + r, gc = k0 + c2;
        unsigned int v = 0u;
        if (gr < M && gc < K) v = *(const unsigned int*)&A[(size_t)gr * K + gc];
        *(unsigned int*)&As[r * ASTR + c2] = v;
      }
#pragma unroll
      for (int it = 0; it < 4; ++it) {
        int e = tid + it * 256;
        int kk = e >> 5, n2 = (e & 31) * 2;
        int gk = k0 + kk, gn = n0 + n2;
        unsigned int v = 0u;
        if (gk < K && gn < N) v = *(const unsigned int*)&W[(size_t)gk * N + gn];
        Bs[n2 * 34 + kk]       = (unsigned short)(v & 0xFFFFu);
        Bs[(n2 + 1) * 34 + kk] = (unsigned short)(v >> 16);
      }
    }
    __syncthreads();

    FragBF af;
    const int row = wave * 16 + l15;
#pragma unroll
    for (int j = 0; j < 8; ++j)
      af.w[j] = *(const unsigned int*)&As[row * ASTR + aK(j, half)];
#pragma unroll
    for (int nt = 0; nt < 4; ++nt) {
      FragBF bf;
      const int col = nt * 16 + l15;
#pragma unroll
      for (int j = 0; j < 8; ++j)
        bf.w[j] = *(const unsigned int*)&Bs[col * 34 + half * 16 + 2 * j];
      acc[nt] = __builtin_amdgcn_wmma_f32_16x16x32_bf16(
          false, af.v, false, bf.v, (short)0, acc[nt], false, false);
    }
    __syncthreads();
  }

#pragma unroll
  for (int nt = 0; nt < 4; ++nt)
#pragma unroll
    for (int r = 0; r < 8; ++r) {
      int gr = m0 + wave * 16 + r + 8 * half;
      int gc = n0 + nt * 16 + l15;
      if (gr < M && gc < N) {
        float v = acc[nt][r];
        if (addsrc) v += addsrc[(size_t)gr * N + gc];
        if (Cf) Cf[(size_t)gr * N + gc] = v;
        else    Cb[(size_t)gr * N + gc] = f2bf(v);
      }
    }
}

// ---------------- fused retention (flash-style, causal decay) ----------------
// grid: (S/64, B*H); block 256 = 8 waves, wave (wr = w>>1, wc = w&1)
// Q/K/V bf16 [B,S,H,HD]; output Yn f32 [B,S,DM] with per-head groupnorm applied.
// Q and K 64x64 tiles loaded by TDM (LDS pad 32dw + 1dw -> stride 66).
__global__ __launch_bounds__(256) void k_retention(const unsigned short* __restrict__ Qb,
    const unsigned short* __restrict__ Kb, const unsigned short* __restrict__ Vb,
    const float* __restrict__ gnw, const float* __restrict__ gnb,
    float* __restrict__ Yn) {
  __shared__ unsigned short Qs[64 * 66];   // [s][d]
  __shared__ unsigned short Ks[64 * 66];   // [t][d]
  __shared__ unsigned short Vt[64 * 66];   // [d][t]
  __shared__ unsigned short Ss[64 * 66];   // [s][t]
  __shared__ float Ysh[64 * 64];
  __shared__ float mrow[64], rrow[64];

  const int tid = threadIdx.x;
  const int wave = tid >> 5, lane = tid & 31;
  const int wr = wave >> 1, wc = wave & 1;
  const int l15 = lane & 15, half = lane >> 4;
  const int bh = blockIdx.y;
  const int b = bh / H_, h = bh % H_;
  const int s0 = blockIdx.x * 64;
  const unsigned qsBase = (unsigned)(size_t)Qs;
  const unsigned ksBase = (unsigned)(size_t)Ks;

  // per-head decay: g_h = 1 - exp(linspace(log 1/32, log 1/512, H))
  const float lg0 = logf(1.f / 32.f), lg1 = logf(1.f / 512.f);
  const float g = 1.f - expf(lg0 + (float)h * (lg1 - lg0) / (float)(H_ - 1));
  const float lgam = logf(g);

  // Q tile via TDM, once
  if (wave == 0) {
    tdm_load_2d(qsBase, Qb + ((size_t)(b * S_ + s0)) * DM_ + h * HD_,
                64u, 64u, 64u, 64u, (unsigned)DM_, 4u, 0u);
    tdm_wait();
  }
  __syncthreads();

  // Q A-fragments cached in registers (2 K-steps of 32 over HD=64)
  FragBF qa[2];
  {
    const int row = wr * 16 + l15;
#pragma unroll
    for (int ks = 0; ks < 2; ++ks)
#pragma unroll
      for (int j = 0; j < 8; ++j)
        qa[ks].w[j] = *(const unsigned int*)&Qs[row * 66 + ks * 32 + aK(j, half)];
  }

  v8f yacc[2];
  yacc[0] = v8zero(); yacc[1] = v8zero();

  const int tTiles = blockIdx.x + 1;   // causal: t <= s
  for (int tt = 0; tt < tTiles; ++tt) {
    const int t0 = tt * 64;
    __syncthreads();   // previous iteration's readers of Ks/Vt/Ss done
    if (wave == 0)     // K tile via TDM
      tdm_load_2d(ksBase, Kb + ((size_t)(b * S_ + t0)) * DM_ + h * HD_,
                  64u, 64u, 64u, 64u, (unsigned)DM_, 4u, 0u);
    // V tile transposed (manual scatter), overlaps the TDM
#pragma unroll
    for (int it = 0; it < 8; ++it) {
      int e = tid + it * 256;
      int r = e >> 5, c2 = (e & 31) * 2;
      unsigned int vv = *(const unsigned int*)&Vb[((size_t)(b * S_ + t0 + r)) * DM_ + h * HD_ + c2];
      Vt[c2 * 66 + r]       = (unsigned short)(vv & 0xFFFFu);
      Vt[(c2 + 1) * 66 + r] = (unsigned short)(vv >> 16);
    }
    if (wave == 0) tdm_wait();
    __syncthreads();

    // S = Q K^T  (M=64, N=64 t, K=HD)
    v8f sac[2];
    sac[0] = v8zero(); sac[1] = v8zero();
#pragma unroll
    for (int ks = 0; ks < 2; ++ks)
#pragma unroll
      for (int a = 0; a < 2; ++a) {
        FragBF bf;
        const int col = wc * 32 + a * 16 + l15;
#pragma unroll
        for (int j = 0; j < 8; ++j)
          bf.w[j] = *(const unsigned int*)&Ks[col * 66 + ks * 32 + half * 16 + 2 * j];
        sac[a] = __builtin_amdgcn_wmma_f32_16x16x32_bf16(
            false, qa[ks].v, false, bf.v, (short)0, sac[a], false, false);
      }

    // decay mask + bf16 -> LDS
#pragma unroll
    for (int a = 0; a < 2; ++a)
#pragma unroll
      for (int r = 0; r < 8; ++r) {
        int sl = wr * 16 + r + 8 * half;
        int tl = wc * 32 + a * 16 + l15;
        int d = (s0 + sl) - (t0 + tl);
        float v = (d >= 0) ? sac[a][r] * __expf(lgam * (float)d) : 0.f;
        Ss[sl * 66 + tl] = f2bf(v);
      }
    __syncthreads();

    // Y += S V  (M=64, N=HD, K=64 t)
#pragma unroll
    for (int ks = 0; ks < 2; ++ks) {
      FragBF sa;
      const int row = wr * 16 + l15;
#pragma unroll
      for (int j = 0; j < 8; ++j)
        sa.w[j] = *(const unsigned int*)&Ss[row * 66 + ks * 32 + aK(j, half)];
#pragma unroll
      for (int a = 0; a < 2; ++a) {
        FragBF bf;
        const int col = wc * 32 + a * 16 + l15;
#pragma unroll
        for (int j = 0; j < 8; ++j)
          bf.w[j] = *(const unsigned int*)&Vt[col * 66 + ks * 32 + half * 16 + 2 * j];
        yacc[a] = __builtin_amdgcn_wmma_f32_16x16x32_bf16(
            false, sa.v, false, bf.v, (short)0, yacc[a], false, false);
      }
    }
  }

  // fused per-head GroupNorm over HD=64
  __syncthreads();
#pragma unroll
  for (int a = 0; a < 2; ++a)
#pragma unroll
    for (int r = 0; r < 8; ++r) {
      int sl = wr * 16 + r + 8 * half;
      int dl = wc * 32 + a * 16 + l15;
      Ysh[sl * 64 + dl] = yacc[a][r];
    }
  __syncthreads();
  if (tid < 64) {
    float s1 = 0.f, s2 = 0.f;
    for (int d = 0; d < 64; ++d) { float v = Ysh[tid * 64 + d]; s1 += v; s2 += v * v; }
    float m = s1 * (1.f / 64.f);
    float var = s2 * (1.f / 64.f) - m * m;
    mrow[tid] = m; rrow[tid] = rsqrtf(var + 1e-5f);
  }
  __syncthreads();
#pragma unroll
  for (int it = 0; it < 16; ++it) {
    int e = tid + it * 256;
    int r = e >> 6, d = e & 63;
    int ch = h * HD_ + d;
    float v = (Ysh[e] - mrow[r]) * rrow[r] * gnw[ch] + gnb[ch];
    Yn[((size_t)(b * S_ + s0 + r)) * DM_ + ch] = v;
  }
}

// ---------------- host ----------------
static inline size_t alignup(size_t x) { return (x + 255) & ~(size_t)255; }
static inline unsigned g1(long n) { return (unsigned)((n + 255) / 256); }

extern "C" void kernel_launch(void* const* d_in, const int* in_sizes, int n_in,
                              void* d_out, int out_size, void* d_ws, size_t ws_size,
                              hipStream_t stream) {
  (void)in_sizes; (void)n_in; (void)out_size; (void)ws_size;
  const float* X      = (const float*)d_in[0];
  const float* wq     = (const float*)d_in[4];
  const float* wk     = (const float*)d_in[5];
  const float* wv     = (const float*)d_in[6];
  const float* wg     = (const float*)d_in[7];
  const float* wo     = (const float*)d_in[8];
  const float* gn_w   = (const float*)d_in[9];
  const float* gn_b   = (const float*)d_in[10];
  const float* ln1_w  = (const float*)d_in[11];
  const float* ln1_b  = (const float*)d_in[12];
  const float* ln2_w  = (const float*)d_in[13];
  const float* ln2_b  = (const float*)d_in[14];
  const float* sglu_u = (const float*)d_in[15];
  const float* sglu_v = (const float*)d_in[16];
  const float* sglu_o = (const float*)d_in[17];
  const float* tp_w1  = (const float*)d_in[18];
  const float* tp_b1  = (const float*)d_in[19];
  const float* tp_w2  = (const float*)d_in[20];
  const float* tp_b2  = (const float*)d_in[21];
  float* out = (float*)d_out;

  const int nW  = ENC_  * DM_;          // 441344  (unpadded-K weights, K=DM)
  const int nWP = ENCP_ * DM_;          // 442368  (K padded to 864)
  char* ws = (char*)d_ws;
  size_t off = 0;
  auto alloc = [&](size_t bytes) { void* p = ws + off; off = alignup(off + bytes); return p; };

  unsigned short* WQb  = (unsigned short*)alloc((size_t)L_ * nWP * 2);
  unsigned short* WKb  = (unsigned short*)alloc((size_t)L_ * nWP * 2);
  unsigned short* WVb  = (unsigned short*)alloc((size_t)L_ * nWP * 2);
  unsigned short* WGb  = (unsigned short*)alloc((size_t)L_ * nWP * 2);
  unsigned short* SUb  = (unsigned short*)alloc((size_t)L_ * nWP * 2);
  unsigned short* SVb  = (unsigned short*)alloc((size_t)L_ * nWP * 2);
  unsigned short* WOb  = (unsigned short*)alloc((size_t)L_ * nW * 2);
  unsigned short* SOb  = (unsigned short*)alloc((size_t)L_ * nW * 2);
  unsigned short* T1b  = (unsigned short*)alloc((size_t)S_ * DFF_ * 2);
  unsigned short* T2b  = (unsigned short*)alloc((size_t)DFF_ * P_ * 2);

  const size_t nAct  = (size_t)BSROWS * ENC_;     // 14,123,008
  const size_t nActP = (size_t)BSROWS * ENCP_;    // 14,155,776
  const size_t nDM   = (size_t)BSROWS * DM_;      //  8,388,608
  float*          Xcur = (float*)alloc(nAct * 4);
  float*          Ybuf = (float*)alloc(nAct * 4);          // also H1 f32 chunk
  unsigned short* Abf  = (unsigned short*)alloc(nActP * 2); // LN out (K-padded) / Xt
  unsigned short* Qb   = (unsigned short*)alloc(nDM * 2);
  unsigned short* Kb   = (unsigned short*)alloc(nDM * 2);
  unsigned short* Vb   = (unsigned short*)alloc(nDM * 2);
  float*          Yn   = (float*)alloc(nDM * 4);           // also Zv / H1b bf16 chunk
  float*          Gf   = (float*)alloc(nDM * 4);           // also Zu / H2 f32 chunk
  unsigned short* Gate = (unsigned short*)alloc(nDM * 2);  // also SGLU product

  // ---- weight conversion (every call; deterministic) ----
  for (int l = 0; l < L_; ++l) {
    const size_t wOff = (size_t)l * H_ * ENC_ * HD_;
    k_cvt_qkvw<<<g1(nWP), 256, 0, stream>>>(wq + wOff, WQb + (size_t)l * nWP, nWP);
    k_cvt_qkvw<<<g1(nWP), 256, 0, stream>>>(wk + wOff, WKb + (size_t)l * nWP, nWP);
    k_cvt_qkvw<<<g1(nWP), 256, 0, stream>>>(wv + wOff, WVb + (size_t)l * nWP, nWP);
    k_cvt_pad <<<g1(nWP), 256, 0, stream>>>(wg + (size_t)l * nW, WGb + (size_t)l * nWP, ENC_, DM_, ENCP_);
    k_cvt_pad <<<g1(nWP), 256, 0, stream>>>(sglu_u + (size_t)l * nW, SUb + (size_t)l * nWP, ENC_, DM_, ENCP_);
    k_cvt_pad <<<g1(nWP), 256, 0, stream>>>(sglu_v + (size_t)l * nW, SVb + (size_t)l * nWP, ENC_, DM_, ENCP_);
    k_cvt_bf16<<<g1(nW), 256, 0, stream>>>(wo + (size_t)l * nW, WOb + (size_t)l * nW, nW);
    k_cvt_bf16<<<g1(nW), 256, 0, stream>>>(sglu_o + (size_t)l * nW, SOb + (size_t)l * nW, nW);
  }
  k_cvt_bf16<<<g1(S_ * DFF_), 256, 0, stream>>>(tp_w1, T1b, S_ * DFF_);
  k_cvt_bf16<<<g1(DFF_ * P_), 256, 0, stream>>>(tp_w2, T2b, DFF_ * P_);

  (void)hipMemcpyAsync(Xcur, X, nAct * 4, hipMemcpyDeviceToDevice, stream);

  const dim3 gDM((DM_ + 63) / 64, (BSROWS + 127) / 128);    // N=512
  const dim3 gENC((ENC_ + 63) / 64, (BSROWS + 127) / 128);  // N=862
  const dim3 gRet(S_ / 64, B_ * H_);

  for (int l = 0; l < L_; ++l) {
    // ln1 -> Abf (bf16, K padded to 864 with zeros)
    k_layernorm<<<BSROWS, 256, 0, stream>>>(Xcur, ln1_w + l * ENC_, ln1_b + l * ENC_, Abf, ENC_, ENCP_);
    // Q,K,V,G projections (K=864, guard-free fast path)
    k_gemm<<<gDM, 256, 0, stream>>>(Abf, WQb + (size_t)l * nWP, nullptr, Qb, nullptr, BSROWS, DM_, ENCP_);
    k_gemm<<<gDM, 256, 0, stream>>>(Abf, WKb + (size_t)l * nWP, nullptr, Kb, nullptr, BSROWS, DM_, ENCP_);
    k_gemm<<<gDM, 256, 0, stream>>>(Abf, WVb + (size_t)l * nWP, nullptr, Vb, nullptr, BSROWS, DM_, ENCP_);
    k_gemm<<<gDM, 256, 0, stream>>>(Abf, WGb + (size_t)l * nWP, Gf, nullptr, nullptr, BSROWS, DM_, ENCP_);
    // fused retention + groupnorm
    k_retention<<<gRet, 256, 0, stream>>>(Qb, Kb, Vb, gn_w + l * DM_, gn_b + l * DM_, Yn);
    // swish gate
    k_gate<<<g1((long)nDM), 256, 0, stream>>>(Gf, Yn, Gate, (int)nDM);
    // wo projection + residual X -> Ybuf
    k_gemm<<<gENC, 256, 0, stream>>>(Gate, WOb + (size_t)l * nW, Ybuf, nullptr, Xcur, BSROWS, ENC_, DM_);
    // ln2 -> Abf
    k_layernorm<<<BSROWS, 256, 0, stream>>>(Ybuf, ln2_w + l * ENC_, ln2_b + l * ENC_, Abf, ENC_, ENCP_);
    // SGLU: Zu, Zv, product, output + residual Y -> Xcur
    k_gemm<<<gDM, 256, 0, stream>>>(Abf, SUb + (size_t)l * nWP, Gf, nullptr, nullptr, BSROWS, DM_, ENCP_);
    k_gemm<<<gDM, 256, 0, stream>>>(Abf, SVb + (size_t)l * nWP, Yn, nullptr, nullptr, BSROWS, DM_, ENCP_);
    k_mul<<<g1((long)nDM), 256, 0, stream>>>(Yn, Gf, Gate, (int)nDM);
    k_gemm<<<gENC, 256, 0, stream>>>(Gate, SOb + (size_t)l * nW, Xcur, nullptr, Ybuf, BSROWS, ENC_, DM_);
  }

  // ---- temporal projection head, chunked over batch (4 x 8) ----
  k_transpose_cvt<<<g1((long)nAct), 256, 0, stream>>>(Xcur, Abf, (int)nAct);  // Xt [B,ENC,S] bf16
  const int BC = 8;
  const int Mc = BC * ENC_;                         // 6896
  const int nH1 = Mc * DFF_;                        // 14,123,008
  const int nH2 = Mc * P_;                          // 662,016
  unsigned short* H1b = (unsigned short*)Yn;        // reuse (28.2MB in 33.6MB)
  float* H2 = Gf;                                   // reuse
  const dim3 gT1((DFF_ + 63) / 64, (Mc + 127) / 128);
  const dim3 gT2((P_ + 63) / 64, (Mc + 127) / 128);
  for (int c = 0; c < B_ / BC; ++c) {
    const unsigned short* Xt = Abf + (size_t)c * BC * ENC_ * S_;
    k_gemm<<<gT1, 256, 0, stream>>>(Xt, T1b, Ybuf, nullptr, nullptr, Mc, DFF_, S_);
    k_gelu_bias<<<g1(nH1), 256, 0, stream>>>(Ybuf, tp_b1, H1b, nH1, DFF_);
    k_gemm<<<gT2, 256, 0, stream>>>(H1b, T2b, H2, nullptr, nullptr, Mc, P_, DFF_);
    k_out_bias_tr<<<g1(nH2), 256, 0, stream>>>(H2, tp_b2, out + (size_t)c * BC * P_ * ENC_, nH2);
  }
}
